// MambaResBlock_6468220748463
// MI455X (gfx1250) — matrix operand order repaired and checked by reading, not compiled, not run on hardware
//
#include <hip/hip_runtime.h>
#include <hip/hip_bf16.h>

// ---------------- problem constants ----------------
#define D_MODEL 1024
#define D_STATE 16
#define D_CONV  4
#define D_INNER 2048
#define DT_RANK 64
#define BATCH   2
#define SEQ     2048
#define MROWS   (BATCH * SEQ)            // 4096 rows through every GEMM
#define XZ_N    (2 * D_INNER)            // 4096
#define DBC_N   (DT_RANK + 2 * D_STATE)  // 96
#define NCHUNK  16                       // scan chunks
#define CHUNK_T (SEQ / NCHUNK)           // 128

typedef __attribute__((ext_vector_type(16))) _Float16 v16h;
typedef __attribute__((ext_vector_type(8)))  _Float16 v8h;
typedef __attribute__((ext_vector_type(8)))  float    v8f;

// ---------------- CDNA5 async copy helpers ----------------
__device__ __forceinline__ void async_ld16(unsigned lds_off, const void* gaddr) {
  // 16B global -> LDS, tracked by ASYNCcnt (ISA 15.18.3 opcode 98)
  asm volatile("global_load_async_to_lds_b128 %0, %1, off"
               :: "v"(lds_off), "v"(gaddr) : "memory");
}
__device__ __forceinline__ void wait_async_prev() {  // allow the 6 just-issued
  asm volatile("s_wait_asynccnt 0x6" ::: "memory");
}
__device__ __forceinline__ void wait_async_all() {
  asm volatile("s_wait_asynccnt 0x0" ::: "memory");
}

// ---------------- LayerNorm -> f16 ----------------
__global__ void ln_kernel(const float* __restrict__ x, const float* __restrict__ w,
                          const float* __restrict__ b, _Float16* __restrict__ xn16) {
  int row = blockIdx.x;
  const float* xr = x + (size_t)row * D_MODEL;
  float v[4];
  float s = 0.f, s2 = 0.f;
#pragma unroll
  for (int j = 0; j < 4; ++j) {
    v[j] = xr[threadIdx.x + 256 * j];
    s += v[j]; s2 += v[j] * v[j];
  }
#pragma unroll
  for (int o = 16; o > 0; o >>= 1) { s += __shfl_xor(s, o, 32); s2 += __shfl_xor(s2, o, 32); }
  __shared__ float sh[8], sh2[8];
  if ((threadIdx.x & 31) == 0) { sh[threadIdx.x >> 5] = s; sh2[threadIdx.x >> 5] = s2; }
  __syncthreads();
  s = 0.f; s2 = 0.f;
#pragma unroll
  for (int j = 0; j < 8; ++j) { s += sh[j]; s2 += sh2[j]; }
  float mu  = s * (1.f / D_MODEL);
  float var = s2 * (1.f / D_MODEL) - mu * mu;
  float rstd = rsqrtf(var + 1e-5f);
#pragma unroll
  for (int j = 0; j < 4; ++j) {
    int c = threadIdx.x + 256 * j;
    xn16[(size_t)row * D_MODEL + c] = (_Float16)((v[j] - mu) * rstd * w[c] + b[c]);
  }
}

// ---------------- f32 (K x N) -> f16 transposed (N x K) ----------------
__global__ void transpose_to_f16(const float* __restrict__ w, _Float16* __restrict__ wt,
                                 int K, int N) {
  size_t idx = (size_t)blockIdx.x * blockDim.x + threadIdx.x;
  if (idx >= (size_t)K * N) return;
  int n = (int)(idx % N);
  int k = (int)(idx / N);
  wt[(size_t)n * K + k] = (_Float16)w[idx];
}

// ---------------- WMMA GEMM with async-LDS double buffering ----------------
// C(MxN,f32) = A(MxK,f16) * Bt(NxK,f16)^T
// Block: 256 threads = 8 waves (2x4). Block tile 128(M) x 256(N) x 32(K).
// Wave tile 64x64 = 4x4 WMMA tiles -> 16 v_wmma per K-step.
// LDS per buffer: A 128 rows + B 256 rows, 40 halves/row (80B stride pad). 2 buffers = 60KB.
#define LROW   40                    // halves per LDS row (32 data + 8 pad)
#define LDS_A_ROWS 128
#define LDS_B_ROWS 256
#define LDS_BUF (LDS_A_ROWS + LDS_B_ROWS) * LROW   // halves per buffer = 15360
__global__ __launch_bounds__(256)
void gemm_wmma_f16(const _Float16* __restrict__ A, const _Float16* __restrict__ Bt,
                   float* __restrict__ C, int M, int N, int K) {
  __shared__ __align__(16) _Float16 smem[2 * LDS_BUF];  // 60 KB
  const unsigned lds_base = (unsigned)(size_t)smem;

  const int tid  = threadIdx.x;
  const int wave = tid >> 5;
  const int lane = tid & 31;
  const int wm = wave >> 2;          // 0..1
  const int wn = wave & 3;           // 0..3
  const int m0b = blockIdx.y * 128;
  const int n0b = blockIdx.x * 256;

  // ISA 16-bit A frag (16x32): lane<16 -> K {kb..kb+7, kb+16..kb+23}, kb=(lane>=16)?8:0
  const int arow = lane & 15;
  const int akb  = (lane & 16) ? 8 : 0;
  // ISA 16-bit B frag (32x16): lane holds column; K {kb..kb+7, kb+16..kb+23}, kb=(lane>=16)?8:0
  const int bcol = lane & 15;
  const int bkb  = (lane & 16) ? 8 : 0;

  v8f acc[4][4];
#pragma unroll
  for (int i = 0; i < 4; ++i)
#pragma unroll
    for (int j = 0; j < 4; ++j) acc[i][j] = (v8f){0.f,0.f,0.f,0.f,0.f,0.f,0.f,0.f};

  // cooperative async copy: A tile 128x32 (512 chunks) + B tile 256x32 (1024 chunks),
  // 16B chunks, 6 per thread
  auto issue_tile = [&](int k0, int p) {
    const unsigned pbase = lds_base + 2u * (unsigned)(p * LDS_BUF);
#pragma unroll
    for (int q = 0; q < 2; ++q) {                 // A chunks
      int cch = q * 256 + tid;                    // 0..511
      int row = cch >> 2, seg = cch & 3;
      const _Float16* ga = A + (size_t)(m0b + row) * K + k0 + seg * 8;
      async_ld16(pbase + 2u * (unsigned)(row * LROW + seg * 8), ga);
    }
#pragma unroll
    for (int q = 0; q < 4; ++q) {                 // B chunks
      int cch = q * 256 + tid;                    // 0..1023
      int row = cch >> 2, seg = cch & 3;
      int nr = n0b + row; if (nr >= N) nr = N - 1;   // clamp loads; stores guarded
      const _Float16* gb = Bt + (size_t)nr * K + k0 + seg * 8;
      async_ld16(pbase + 2u * (unsigned)((LDS_A_ROWS + row) * LROW + seg * 8), gb);
    }
  };

  const int nk = K >> 5;
  issue_tile(0, 0);
  for (int it = 0; it < nk; ++it) {
    const int p = it & 1;
    if (it + 1 < nk) { issue_tile((it + 1) << 5, p ^ 1); wait_async_prev(); }
    else             { wait_async_all(); }
    __syncthreads();   // tile `it` visible to all waves

    const _Float16* sa = smem + p * LDS_BUF;
    const _Float16* sb = sa + LDS_A_ROWS * LROW;
    v16h af[4], bf[4];
#pragma unroll
    for (int i = 0; i < 4; ++i) {
      const _Float16* ap = &sa[(wm * 64 + 16 * i + arow) * LROW + akb];
      v8h lo = *(const v8h*)(ap);
      v8h hi = *(const v8h*)(ap + 16);
      af[i] = __builtin_shufflevector(lo, hi, 0,1,2,3,4,5,6,7,8,9,10,11,12,13,14,15);
    }
#pragma unroll
    for (int j = 0; j < 4; ++j) {
      const _Float16* bp = &sb[(wn * 64 + 16 * j + bcol) * LROW + bkb];
      v8h lo = *(const v8h*)(bp);
      v8h hi = *(const v8h*)(bp + 16);
      bf[j] = __builtin_shufflevector(lo, hi, 0,1,2,3,4,5,6,7,8,9,10,11,12,13,14,15);
    }
#pragma unroll
    for (int i = 0; i < 4; ++i)
#pragma unroll
      for (int j = 0; j < 4; ++j)
        acc[i][j] = __builtin_amdgcn_wmma_f32_16x16x32_f16(
            false, af[i], false, bf[j], (short)0, acc[i][j], false, false);

    if (it + 1 < nk) __syncthreads();  // protect buffer p from next-next tile's async writes
  }

  // C/D layout: VGPR r, lane<16 -> M=r, lane>=16 -> M=8+r; N = lane&15
  const int crh  = (lane & 16) ? 8 : 0;
  const int ccol = lane & 15;
#pragma unroll
  for (int i = 0; i < 4; ++i)
#pragma unroll
    for (int j = 0; j < 4; ++j) {
      int n = n0b + wn * 64 + 16 * j + ccol;
      if (n >= N) continue;
      float* cp = C + (size_t)(m0b + wm * 64 + 16 * i + crh) * N + n;
#pragma unroll
      for (int r = 0; r < 8; ++r) cp[(size_t)r * N] = acc[i][j][r];
    }
}

// ---------------- depthwise causal conv(4) + bias + SiLU ----------------
__global__ void conv_silu_kernel(const float* __restrict__ xz, const float* __restrict__ cw,
                                 const float* __restrict__ cb, float* __restrict__ xc,
                                 _Float16* __restrict__ xc16) {
  size_t idx = (size_t)blockIdx.x * blockDim.x + threadIdx.x;
  if (idx >= (size_t)MROWS * D_INNER) return;
  int d = (int)(idx % D_INNER);
  size_t bt = idx / D_INNER;           // b*SEQ + t
  int t = (int)(bt % SEQ);
  float acc = cb[d];
#pragma unroll
  for (int j = 0; j < D_CONV; ++j) {
    int tt = t - (D_CONV - 1) + j;
    if (tt >= 0)
      acc += cw[d * D_CONV + j] * xz[(bt - (D_CONV - 1) + j) * XZ_N + d];
  }
  float s = acc / (1.f + expf(-acc));  // SiLU
  xc[idx] = s;
  xc16[idx] = (_Float16)s;
}

// ---------------- dbc[:, :DT_RANK] -> f16 ----------------
__global__ void dt_slice_kernel(const float* __restrict__ dbc, _Float16* __restrict__ dt16) {
  size_t idx = (size_t)blockIdx.x * blockDim.x + threadIdx.x;
  if (idx >= (size_t)MROWS * DT_RANK) return;
  size_t r = idx / DT_RANK;
  int c = (int)(idx % DT_RANK);
  dt16[idx] = (_Float16)dbc[r * DBC_N + c];
}

// ---------------- softplus(dt + bias) in-place ----------------
__global__ void softplus_kernel(float* __restrict__ dtv, const float* __restrict__ bias) {
  size_t idx = (size_t)blockIdx.x * blockDim.x + threadIdx.x;
  if (idx >= (size_t)MROWS * D_INNER) return;
  int d = (int)(idx % D_INNER);
  float v = dtv[idx] + bias[d];
  dtv[idx] = (v > 20.f) ? v : log1pf(expf(v));
}

// ---------------- chunked selective scan ----------------
// Linear diagonal recurrence h_t = e_t*h + g_t composes exactly across chunks:
// chunk summary E[s] = exp(a[s] * sum_t dt_t), H[s] = scan from zero state.
// Index map for pass1/3: idx -> d (low 11 bits), c (next 4), b.
// Storage layout: X[((b*NCHUNK + c)*16 + s)*D_INNER + d]  (lane-coalesced in d)

__global__ void scan_pass1(const float* __restrict__ xc, const float* __restrict__ dtv,
                           const float* __restrict__ dbc, const float* __restrict__ A_log,
                           float* __restrict__ Ec, float* __restrict__ Hc) {
  int idx = blockIdx.x * blockDim.x + threadIdx.x;
  int d = idx & (D_INNER - 1);
  int rest = idx >> 11;
  int c = rest & (NCHUNK - 1);
  int b = rest >> 4;
  float a[D_STATE], h[D_STATE];
#pragma unroll
  for (int s = 0; s < D_STATE; ++s) { a[s] = -expf(A_log[d * D_STATE + s]); h[s] = 0.f; }
  float Sdt = 0.f;
  int t0 = c * CHUNK_T;
  for (int t = t0; t < t0 + CHUNK_T; ++t) {
    size_t bt = (size_t)b * SEQ + t;
    float u   = xc[bt * D_INNER + d];
    float dtt = dtv[bt * D_INNER + d];
    const float* bc = dbc + bt * DBC_N;   // uniform across wave -> broadcast
    float du = dtt * u;
    Sdt += dtt;
#pragma unroll
    for (int s = 0; s < D_STATE; ++s)
      h[s] = expf(dtt * a[s]) * h[s] + du * bc[DT_RANK + s];
  }
  size_t o0 = ((size_t)(b * NCHUNK + c) * D_STATE) * D_INNER + d;
#pragma unroll
  for (int s = 0; s < D_STATE; ++s) {
    Ec[o0 + (size_t)s * D_INNER] = expf(a[s] * Sdt);
    Hc[o0 + (size_t)s * D_INNER] = h[s];
  }
}

// serial over the NCHUNK chunk summaries; overwrites Ec with each chunk's initial state
__global__ void scan_pass2(float* __restrict__ Ec, float* __restrict__ Hc) {
  int idx = blockIdx.x * blockDim.x + threadIdx.x;   // 0 .. B*D_INNER-1
  int d = idx & (D_INNER - 1);
  int b = idx >> 11;
  float carry[D_STATE];
#pragma unroll
  for (int s = 0; s < D_STATE; ++s) carry[s] = 0.f;
  for (int c = 0; c < NCHUNK; ++c) {
    size_t o0 = ((size_t)(b * NCHUNK + c) * D_STATE) * D_INNER + d;
#pragma unroll
    for (int s = 0; s < D_STATE; ++s) {
      size_t o = o0 + (size_t)s * D_INNER;
      float E = Ec[o], H = Hc[o];
      Ec[o] = carry[s];                 // h_init for chunk c
      carry[s] = E * carry[s] + H;
    }
  }
}

__global__ void scan_pass3(const float* __restrict__ xc, const float* __restrict__ dtv,
                           const float* __restrict__ dbc, const float* __restrict__ xz,
                           const float* __restrict__ A_log, const float* __restrict__ Dsk,
                           const float* __restrict__ Hinit, _Float16* __restrict__ y16) {
  int idx = blockIdx.x * blockDim.x + threadIdx.x;
  int d = idx & (D_INNER - 1);
  int rest = idx >> 11;
  int c = rest & (NCHUNK - 1);
  int b = rest >> 4;
  float a[D_STATE], h[D_STATE];
  size_t o0 = ((size_t)(b * NCHUNK + c) * D_STATE) * D_INNER + d;
#pragma unroll
  for (int s = 0; s < D_STATE; ++s) {
    a[s] = -expf(A_log[d * D_STATE + s]);
    h[s] = Hinit[o0 + (size_t)s * D_INNER];
  }
  float dskip = Dsk[d];
  int t0 = c * CHUNK_T;
  for (int t = t0; t < t0 + CHUNK_T; ++t) {
    size_t bt = (size_t)b * SEQ + t;
    float u   = xc[bt * D_INNER + d];
    float dtt = dtv[bt * D_INNER + d];
    const float* bc = dbc + bt * DBC_N;
    float du = dtt * u;
    float y = 0.f;
#pragma unroll
    for (int s = 0; s < D_STATE; ++s) {
      h[s] = expf(dtt * a[s]) * h[s] + du * bc[DT_RANK + s];
      y += h[s] * bc[DT_RANK + D_STATE + s];
    }
    y += u * dskip;
    float z = xz[bt * XZ_N + D_INNER + d];
    float sz = z / (1.f + expf(-z));
    y16[bt * D_INNER + d] = (_Float16)(y * sz);
  }
}

// ---------------- residual add ----------------
__global__ void add_res_kernel(const float* __restrict__ x, const float* __restrict__ g,
                               float* __restrict__ out) {
  size_t idx = (size_t)blockIdx.x * blockDim.x + threadIdx.x;
  if (idx >= (size_t)MROWS * D_MODEL) return;
  out[idx] = x[idx] + g[idx];
}

// ---------------- host orchestration ----------------
extern "C" void kernel_launch(void* const* d_in, const int* in_sizes, int n_in,
                              void* d_out, int out_size, void* d_ws, size_t ws_size,
                              hipStream_t stream) {
  const float* x         = (const float*)d_in[0];
  const float* ln_w      = (const float*)d_in[1];
  const float* ln_b      = (const float*)d_in[2];
  const float* in_proj_w = (const float*)d_in[3];   // (1024, 4096)
  const float* conv_w    = (const float*)d_in[4];   // (2048, 4)
  const float* conv_b    = (const float*)d_in[5];
  const float* x_proj_w  = (const float*)d_in[6];   // (2048, 96)
  const float* dt_proj_w = (const float*)d_in[7];   // (64, 2048)
  const float* dt_proj_b = (const float*)d_in[8];
  const float* A_log     = (const float*)d_in[9];   // (2048, 16)
  const float* D_skip    = (const float*)d_in[10];
  const float* out_proj_w= (const float*)d_in[11];  // (2048, 1024)
  float* out = (float*)d_out;

  char* base = (char*)d_ws;
  size_t off = 0;
  auto take = [&](size_t bytes) -> char* {
    char* p = base + off;
    off = (off + bytes + 255) & ~(size_t)255;
    return p;
  };
  _Float16* xn16 = (_Float16*)take((size_t)MROWS * D_MODEL * 2);
  _Float16* w_in = (_Float16*)take((size_t)XZ_N * D_MODEL * 2);    // (4096,1024) = in_proj^T
  _Float16* w_xp = (_Float16*)take((size_t)DBC_N * D_INNER * 2);   // (96,2048)
  _Float16* w_dt = (_Float16*)take((size_t)D_INNER * DT_RANK * 2); // (2048,64)
  _Float16* w_out= (_Float16*)take((size_t)D_MODEL * D_INNER * 2); // (1024,2048)
  float*    xz   = (float*)take((size_t)MROWS * XZ_N * 4);
  float*    xc   = (float*)take((size_t)MROWS * D_INNER * 4);
  _Float16* xc16 = (_Float16*)take((size_t)MROWS * D_INNER * 2);
  float*    dbc  = (float*)take((size_t)MROWS * DBC_N * 4);
  _Float16* dt16 = (_Float16*)take((size_t)MROWS * DT_RANK * 2);
  float*    dtv  = (float*)take((size_t)MROWS * D_INNER * 4);
  _Float16* y16  = (_Float16*)take((size_t)MROWS * D_INNER * 2);
  float*    gout = (float*)take((size_t)MROWS * D_MODEL * 4);
  float*    Ec   = (float*)take((size_t)BATCH * NCHUNK * D_STATE * D_INNER * 4);
  float*    Hc   = (float*)take((size_t)BATCH * NCHUNK * D_STATE * D_INNER * 4);

  // 1. LayerNorm -> f16
  ln_kernel<<<MROWS, 256, 0, stream>>>(x, ln_w, ln_b, xn16);

  // 2. transpose+convert weights to f16 (N x K)
  {
    size_t n;
    n = (size_t)D_MODEL * XZ_N;
    transpose_to_f16<<<(unsigned)((n + 255) / 256), 256, 0, stream>>>(in_proj_w, w_in, D_MODEL, XZ_N);
    n = (size_t)D_INNER * DBC_N;
    transpose_to_f16<<<(unsigned)((n + 255) / 256), 256, 0, stream>>>(x_proj_w, w_xp, D_INNER, DBC_N);
    n = (size_t)DT_RANK * D_INNER;
    transpose_to_f16<<<(unsigned)((n + 255) / 256), 256, 0, stream>>>(dt_proj_w, w_dt, DT_RANK, D_INNER);
    n = (size_t)D_INNER * D_MODEL;
    transpose_to_f16<<<(unsigned)((n + 255) / 256), 256, 0, stream>>>(out_proj_w, w_out, D_INNER, D_MODEL);
  }

  // 3. xz = xn @ in_proj_w   (4096 x 4096 x 1024), block tile 128x256
  gemm_wmma_f16<<<dim3(XZ_N / 256, MROWS / 128), 256, 0, stream>>>(xn16, w_in, xz, MROWS, XZ_N, D_MODEL);

  // 4. depthwise conv + SiLU -> xc, xc16
  {
    size_t n = (size_t)MROWS * D_INNER;
    conv_silu_kernel<<<(unsigned)((n + 255) / 256), 256, 0, stream>>>(xz, conv_w, conv_b, xc, xc16);
  }

  // 5. dbc = xc @ x_proj_w   (4096 x 96 x 2048)
  gemm_wmma_f16<<<dim3((DBC_N + 255) / 256, MROWS / 128), 256, 0, stream>>>(xc16, w_xp, dbc, MROWS, DBC_N, D_INNER);

  // 6. dt16 = f16(dbc[:, :64])
  {
    size_t n = (size_t)MROWS * DT_RANK;
    dt_slice_kernel<<<(unsigned)((n + 255) / 256), 256, 0, stream>>>(dbc, dt16);
  }

  // 7. dtv = dt16 @ dt_proj_w   (4096 x 2048 x 64)
  gemm_wmma_f16<<<dim3(D_INNER / 256, MROWS / 128), 256, 0, stream>>>(dt16, w_dt, dtv, MROWS, D_INNER, DT_RANK);

  // 8. dtv = softplus(dtv + dt_proj_b)
  {
    size_t n = (size_t)MROWS * D_INNER;
    softplus_kernel<<<(unsigned)((n + 255) / 256), 256, 0, stream>>>(dtv, dt_proj_b);
  }

  // 9. chunked selective scan -> y16
  scan_pass1<<<(BATCH * NCHUNK * D_INNER) / 256, 256, 0, stream>>>(xc, dtv, dbc, A_log, Ec, Hc);
  scan_pass2<<<(BATCH * D_INNER) / 256, 256, 0, stream>>>(Ec, Hc);
  scan_pass3<<<(BATCH * NCHUNK * D_INNER) / 256, 256, 0, stream>>>(xc, dtv, dbc, xz, A_log, D_skip, Ec, y16);

  // 10. gout = y @ out_proj_w   (4096 x 1024 x 2048)
  gemm_wmma_f16<<<dim3(D_MODEL / 256, MROWS / 128), 256, 0, stream>>>(y16, w_out, gout, MROWS, D_MODEL, D_INNER);

  // 11. out = x + gout
  {
    size_t n = (size_t)MROWS * D_MODEL;
    add_res_kernel<<<(unsigned)((n + 255) / 256), 256, 0, stream>>>(x, gout, out);
  }
}
